// MogLSTM_24034636988672
// MI455X (gfx1250) — compile-verified, matrix operand order checked by reading, not run on hardware
//
#include <hip/hip_runtime.h>
#include <hip/hip_bf16.h>
#include <stdint.h>

// Problem constants (B=32, N=64, L=64, C=512, H=512, MOG=5)
#define BN    2048
#define CDIM  512
#define HDIM  512
#define LSEQ  64
#define NGATE 4

typedef __attribute__((ext_vector_type(16))) __bf16 bf16x16;
typedef __attribute__((ext_vector_type(8)))  float  f32x8;

union Frag { uint32_t u[8]; bf16x16 v; };

__device__ __forceinline__ unsigned short f2bf(float f) {
  uint32_t u = __builtin_bit_cast(uint32_t, f);
  u += 0x7fffu + ((u >> 16) & 1u);   // round-to-nearest-even bf16
  return (unsigned short)(u >> 16);
}

__device__ __forceinline__ float sigf(float x) { return 1.0f / (1.0f + __expf(-x)); }

// CDNA5 16-bit A-operand layout (ISA 7.12.2): lane = {half,row}; half 0 holds
// K = [0..7],[16..23], half 1 holds K = [8..15],[24..31]. Each half is two
// contiguous 16-byte runs -> two global_load_b128 per lane. B uses the mirror
// layout (lane<->N) so we feed it from pre-transposed (N-major) weights.
__device__ __forceinline__ void load_frag(const unsigned short* __restrict__ base,
                                          int ld, int row0, int k0, int lane, Frag& f) {
  const int half = lane >> 4;
  const int r    = lane & 15;
  const unsigned short* p = base + (long)(row0 + r) * ld + k0 + half * 8;
  uint4 lo = *reinterpret_cast<const uint4*>(p);       // K = k0 + half*8 .. +7
  uint4 hi = *reinterpret_cast<const uint4*>(p + 16);  // K = k0 +16+ half*8 .. +7
  f.u[0] = lo.x; f.u[1] = lo.y; f.u[2] = lo.z; f.u[3] = lo.w;
  f.u[4] = hi.x; f.u[5] = hi.y; f.u[6] = hi.z; f.u[7] = hi.w;
}

__device__ __forceinline__ f32x8 wmma_bf16(const Frag& a, const Frag& b, f32x8 c) {
  return __builtin_amdgcn_wmma_f32_16x16x32_bf16(false, a.v, false, b.v,
                                                 (short)0, c, false, false);
}

__device__ __forceinline__ f32x8 zero8() {
  f32x8 z = {0.f, 0.f, 0.f, 0.f, 0.f, 0.f, 0.f, 0.f};
  return z;
}

// 32x32 output tile per wave: 2x2 WMMAs per 32-wide K chunk (2.0 b128/WMMA).
__device__ __forceinline__ void gemm32x32(const unsigned short* __restrict__ A, int lda, int m0,
                                          const unsigned short* __restrict__ Bt, int ldb, int n0,
                                          int K, int lane, f32x8 acc[2][2]) {
  for (int k0 = 0; k0 < K; k0 += 32) {
    Frag a0, a1, b0, b1;
    load_frag(A,  lda, m0,      k0, lane, a0);
    load_frag(A,  lda, m0 + 16, k0, lane, a1);
    load_frag(Bt, ldb, n0,      k0, lane, b0);
    load_frag(Bt, ldb, n0 + 16, k0, lane, b1);
    acc[0][0] = wmma_bf16(a0, b0, acc[0][0]);
    acc[0][1] = wmma_bf16(a0, b1, acc[0][1]);
    acc[1][0] = wmma_bf16(a1, b0, acc[1][0]);
    acc[1][1] = wmma_bf16(a1, b1, acc[1][1]);
  }
}

// ---- one-time prep kernels -------------------------------------------------

// Wt[n*K + k] = bf16(W[k*N + n])   (weights -> transposed bf16)
__global__ __launch_bounds__(256) void wconv_kernel(const float* __restrict__ W,
                                                    unsigned short* __restrict__ Wt,
                                                    int K, int N) {
  int idx = blockIdx.x * 256 + threadIdx.x;
  if (idx >= K * N) return;
  int n = idx / K, k = idx - n * K;
  Wt[idx] = f2bf(W[(long)k * N + n]);
}

__global__ __launch_bounds__(256) void init_state_kernel(float* __restrict__ ht,
                                                         float* __restrict__ Ct,
                                                         unsigned short* __restrict__ htb) {
  int i = blockIdx.x * 256 + threadIdx.x;
  ht[i] = 0.0f; Ct[i] = 0.0f; htb[i] = 0;
}

// ---- per-step kernels ------------------------------------------------------

// out = 2*sigmoid(Abf @ Bt^T) * src  ; writes f32 + bf16 shadow. M=BN, N=K=512.
__global__ __launch_bounds__(256) void mog_gemm_kernel(const unsigned short* __restrict__ Abf,
                                                       const unsigned short* __restrict__ Bt,
                                                       const float* __restrict__ src,
                                                       long src_stride,
                                                       float* __restrict__ out_f,
                                                       unsigned short* __restrict__ out_bf) {
  const int lane = threadIdx.x & 31;
  const int wave = threadIdx.x >> 5;
  const int m0 = blockIdx.x * 128 + (wave >> 1) * 32;
  const int n0 = blockIdx.y * 64  + (wave & 1) * 32;
  f32x8 acc[2][2] = {{zero8(), zero8()}, {zero8(), zero8()}};
  gemm32x32(Abf, HDIM, m0, Bt, HDIM, n0, HDIM, lane, acc);
  const int half = lane >> 4, nc = lane & 15;
#pragma unroll
  for (int ti = 0; ti < 2; ++ti)
#pragma unroll
    for (int tj = 0; tj < 2; ++tj) {
      int col = n0 + 16 * tj + nc;
#pragma unroll
      for (int r = 0; r < 8; ++r) {
        int row = m0 + 16 * ti + r + 8 * half;
        float s = 2.0f * sigf(acc[ti][tj][r]);
        float v = s * src[(long)row * src_stride + col];
        long o = (long)row * HDIM + col;
        out_f[o]  = v;
        out_bf[o] = f2bf(v);
      }
    }
}

// mt = xt@Wmx + ht@Wmh  -> bf16 only (consumed only as a GEMM operand)
__global__ __launch_bounds__(256) void mt_kernel(const unsigned short* __restrict__ xtb,
                                                 const unsigned short* __restrict__ htb,
                                                 const unsigned short* __restrict__ Wmxt,
                                                 const unsigned short* __restrict__ Wmht,
                                                 unsigned short* __restrict__ mtb) {
  const int lane = threadIdx.x & 31;
  const int wave = threadIdx.x >> 5;
  const int m0 = blockIdx.x * 128 + (wave >> 1) * 32;
  const int n0 = blockIdx.y * 64  + (wave & 1) * 32;
  f32x8 acc[2][2] = {{zero8(), zero8()}, {zero8(), zero8()}};
  gemm32x32(xtb, CDIM, m0, Wmxt, CDIM, n0, CDIM, lane, acc);
  gemm32x32(htb, HDIM, m0, Wmht, HDIM, n0, HDIM, lane, acc);
  const int half = lane >> 4, nc = lane & 15;
#pragma unroll
  for (int ti = 0; ti < 2; ++ti)
#pragma unroll
    for (int tj = 0; tj < 2; ++tj) {
      int col = n0 + 16 * tj + nc;
#pragma unroll
      for (int r = 0; r < 8; ++r) {
        int row = m0 + 16 * ti + r + 8 * half;
        mtb[(long)row * HDIM + col] = f2bf(acc[ti][tj][r]);
      }
    }
}

// gates = xt@Wih + mt@Whm + bih + bhh computed for all 4 gate column-groups of
// the same 32x32 tile (A fragments reused across N-tiles, reloaded per gate), then
// Ct = sig(fg)*Ct + sig(ig)*tanh(cg); ht = sig(og)*tanh(Ct). The 2048x2048
// gate matrix is never materialized.
__global__ __launch_bounds__(256) void gates_cell_kernel(const unsigned short* __restrict__ xtb,
                                                         const unsigned short* __restrict__ mtb,
                                                         const unsigned short* __restrict__ Wiht,
                                                         const unsigned short* __restrict__ Whmt,
                                                         const float* __restrict__ bih,
                                                         const float* __restrict__ bhh,
                                                         float* __restrict__ Ct,
                                                         float* __restrict__ ht_f,
                                                         unsigned short* __restrict__ htb) {
  const int lane = threadIdx.x & 31;
  const int wave = threadIdx.x >> 5;
  const int m0 = blockIdx.x * 64  + (wave >> 2) * 32;  // 2 waves along M
  const int n0 = blockIdx.y * 128 + (wave & 3) * 32;   // 4 waves along N
  f32x8 acc[NGATE][2][2];
#pragma unroll
  for (int g = 0; g < NGATE; ++g) {
    acc[g][0][0] = zero8(); acc[g][0][1] = zero8();
    acc[g][1][0] = zero8(); acc[g][1][1] = zero8();
  }
  for (int g = 0; g < NGATE; ++g) {
    gemm32x32(xtb, CDIM, m0, Wiht, CDIM, g * HDIM + n0, CDIM, lane, acc[g]);
    gemm32x32(mtb, HDIM, m0, Whmt, HDIM, g * HDIM + n0, HDIM, lane, acc[g]);
  }
  const int half = lane >> 4, nc = lane & 15;
#pragma unroll
  for (int ti = 0; ti < 2; ++ti)
#pragma unroll
    for (int j = 0; j < 2; ++j) {
      int col = n0 + 16 * j + nc;
#pragma unroll
      for (int r = 0; r < 8; ++r) {
        int row = m0 + 16 * ti + r + 8 * half;
        float ig = acc[0][ti][j][r] + bih[col]            + bhh[col];
        float fg = acc[1][ti][j][r] + bih[HDIM + col]     + bhh[HDIM + col];
        float cg = acc[2][ti][j][r] + bih[2 * HDIM + col] + bhh[2 * HDIM + col];
        float og = acc[3][ti][j][r] + bih[3 * HDIM + col] + bhh[3 * HDIM + col];
        long o = (long)row * HDIM + col;
        float cnew = sigf(fg) * Ct[o] + sigf(ig) * tanhf(cg);
        float hnew = sigf(og) * tanhf(cnew);
        Ct[o]   = cnew;
        ht_f[o] = hnew;
        htb[o]  = f2bf(hnew);
      }
    }
}

// out[bn] = ht[bn,:] . Wout + bout
__global__ __launch_bounds__(256) void out_proj_kernel(const float* __restrict__ ht,
                                                       const float* __restrict__ Wout,
                                                       const float* __restrict__ bout,
                                                       float* __restrict__ out) {
  int i = blockIdx.x * 256 + threadIdx.x;
  if (i >= BN) return;
  float s = bout[0];
  const float* row = ht + (long)i * HDIM;
#pragma unroll 4
  for (int h = 0; h < HDIM; ++h) s += row[h] * Wout[h];
  out[i] = s;
}

// ---- host driver -----------------------------------------------------------

extern "C" void kernel_launch(void* const* d_in, const int* in_sizes, int n_in,
                              void* d_out, int out_size, void* d_ws, size_t ws_size,
                              hipStream_t stream) {
  const float* x    = (const float*)d_in[0];   // (B,N,L,C)
  const float* Wih  = (const float*)d_in[1];   // (C,4H)
  const float* bih  = (const float*)d_in[2];   // (4H)
  const float* bhh  = (const float*)d_in[3];   // (4H)
  const float* Wmx  = (const float*)d_in[4];   // (C,H)
  const float* Wmh  = (const float*)d_in[5];   // (H,H)
  const float* Whm  = (const float*)d_in[6];   // (H,4H)
  const float* Q    = (const float*)d_in[7];   // (H,C)
  const float* R    = (const float*)d_in[8];   // (C,H)
  const float* Wout = (const float*)d_in[9];   // (H,1)
  const float* bout = (const float*)d_in[10];  // (1)
  // d_in[11] = mog_iterations (device scalar, == 5; loop structure hardcoded)

  char* ws = (char*)d_ws;
  auto take = [&](size_t bytes) { char* p = ws; ws += (bytes + 255) & ~(size_t)255; return p; };
  const size_t EL = (size_t)BN * HDIM;
  float*          xt_f = (float*)take(EL * 4);
  float*          ht_f = (float*)take(EL * 4);
  float*          Ct   = (float*)take(EL * 4);
  unsigned short* xtb  = (unsigned short*)take(EL * 2);
  unsigned short* htb  = (unsigned short*)take(EL * 2);
  unsigned short* mtb  = (unsigned short*)take(EL * 2);
  unsigned short* Qt   = (unsigned short*)take((size_t)HDIM * CDIM * 2);
  unsigned short* Rt   = (unsigned short*)take((size_t)CDIM * HDIM * 2);
  unsigned short* Wmxt = (unsigned short*)take((size_t)CDIM * HDIM * 2);
  unsigned short* Wmht = (unsigned short*)take((size_t)HDIM * HDIM * 2);
  unsigned short* Wiht = (unsigned short*)take((size_t)CDIM * 4 * HDIM * 2);
  unsigned short* Whmt = (unsigned short*)take((size_t)HDIM * 4 * HDIM * 2);

  const dim3 blk(256);
  // weights -> bf16 transposed (B^T is N-major so B fragments load contiguously)
  wconv_kernel<<<dim3((HDIM * CDIM + 255) / 256), blk, 0, stream>>>(Q,   Qt,   HDIM, CDIM);
  wconv_kernel<<<dim3((CDIM * HDIM + 255) / 256), blk, 0, stream>>>(R,   Rt,   CDIM, HDIM);
  wconv_kernel<<<dim3((CDIM * HDIM + 255) / 256), blk, 0, stream>>>(Wmx, Wmxt, CDIM, HDIM);
  wconv_kernel<<<dim3((HDIM * HDIM + 255) / 256), blk, 0, stream>>>(Wmh, Wmht, HDIM, HDIM);
  wconv_kernel<<<dim3((CDIM * 4 * HDIM + 255) / 256), blk, 0, stream>>>(Wih, Wiht, CDIM, 4 * HDIM);
  wconv_kernel<<<dim3((HDIM * 4 * HDIM + 255) / 256), blk, 0, stream>>>(Whm, Whmt, HDIM, 4 * HDIM);
  init_state_kernel<<<dim3(EL / 256), blk, 0, stream>>>(ht_f, Ct, htb);

  const dim3 gemm_grid(BN / 128, HDIM / 64);   // 16 x 8 blocks, 8 waves each
  const dim3 gate_grid(BN / 64,  HDIM / 128);  // 32 x 4 blocks, 8 waves each

  for (int t = 0; t < LSEQ; ++t) {
    const float* xsrc = x + (long)t * CDIM;    // row bn at stride L*C
    // mogrify i=1: xt = 2*sig(ht@Q) * x[t]
    mog_gemm_kernel<<<gemm_grid, blk, 0, stream>>>(htb, Qt, xsrc, (long)LSEQ * CDIM, xt_f, xtb);
    // i=2: ht = 2*sig(xt@R) * ht
    mog_gemm_kernel<<<gemm_grid, blk, 0, stream>>>(xtb, Rt, ht_f, (long)HDIM, ht_f, htb);
    // i=3: xt = 2*sig(ht@Q) * xt
    mog_gemm_kernel<<<gemm_grid, blk, 0, stream>>>(htb, Qt, xt_f, (long)HDIM, xt_f, xtb);
    // i=4: ht = 2*sig(xt@R) * ht
    mog_gemm_kernel<<<gemm_grid, blk, 0, stream>>>(xtb, Rt, ht_f, (long)HDIM, ht_f, htb);
    // i=5: xt = 2*sig(ht@Q) * xt
    mog_gemm_kernel<<<gemm_grid, blk, 0, stream>>>(htb, Qt, xt_f, (long)HDIM, xt_f, xtb);
    // mt = xt@Wmx + ht@Wmh
    mt_kernel<<<gemm_grid, blk, 0, stream>>>(xtb, htb, Wmxt, Wmht, mtb);
    // gates + cell update (all 4 gates fused, no gate matrix materialized)
    gates_cell_kernel<<<gate_grid, blk, 0, stream>>>(xtb, mtb, Wiht, Whmt, bih, bhh,
                                                     Ct, ht_f, htb);
  }
  out_proj_kernel<<<dim3((BN + 255) / 256), blk, 0, stream>>>(ht_f, Wout, bout, (float*)d_out);
}
